// SelfAttention_17317308137728
// MI455X (gfx1250) — matrix operand order repaired and checked by reading, not compiled, not run on hardware
//
#include <hip/hip_runtime.h>
#include <hip/hip_bf16.h>

typedef __attribute__((ext_vector_type(16))) _Float16 v16h;
typedef __attribute__((ext_vector_type(8)))  _Float16 v8h;
typedef __attribute__((ext_vector_type(8)))  float    v8f;
typedef int v4i __attribute__((vector_size(4 * sizeof(int))));   // matches builtin param

union AFrag { v16h v; v8h h[2]; };

static __device__ __forceinline__ v8f wmma_f16(v16h a, v16h b, v8f c) {
    return __builtin_amdgcn_wmma_f32_16x16x32_f16(false, a, false, b,
                                                  (short)0, c, false, false);
}

// ---- CDNA5 async global->LDS path (guarded, with synchronous fallback) ----
#if defined(__has_builtin)
#  if __has_builtin(__builtin_amdgcn_global_load_async_to_lds_b128)
#    define HAVE_ASYNC_LDS 1
#  endif
#endif
#ifndef HAVE_ASYNC_LDS
#  define HAVE_ASYNC_LDS 0
#endif

#if HAVE_ASYNC_LDS
static __device__ __forceinline__ void async_copy16(const _Float16* g, _Float16* l) {
    __builtin_amdgcn_global_load_async_to_lds_b128(
        (__attribute__((address_space(1))) v4i*)(g),
        (__attribute__((address_space(3))) v4i*)(l), 0, 0);
}
#endif

static __device__ __forceinline__ void wait_async0() {
#if defined(__has_builtin)
#  if __has_builtin(__builtin_amdgcn_s_wait_asynccnt)
    __builtin_amdgcn_s_wait_asynccnt(0);
    return;
#  endif
#endif
    asm volatile("s_wait_asynccnt 0x0" ::: "memory");
}

// ---------------------------------------------------------------------------
// K1: projections. D = W_all(1024x512) * X(512x4096) per batch.
// Rows 0..255 -> F, 256..511 -> G, 512..1023 -> Hm.
// F,G stored transposed f16 [N=4096, HC=256]; Hm row-major f16 [512, 4096].
// One wave computes a 64x16 block (4 M-tiles) so the strided X gather (B-frag)
// is amortized over 4 WMMAs. blockDim=128 (4 waves).
// ---------------------------------------------------------------------------
__global__ void proj_kernel(const float* __restrict__ x,
                            const float* __restrict__ Wf, const float* __restrict__ bf,
                            const float* __restrict__ Wg, const float* __restrict__ bg,
                            const float* __restrict__ Wh, const float* __restrict__ bh,
                            _Float16* __restrict__ Ft, _Float16* __restrict__ Gt,
                            _Float16* __restrict__ Hm)
{
    const int C = 512, N = 4096, HC = 256;
    int gw   = blockIdx.x * 4 + (threadIdx.x >> 5);   // global wave id
    int lane = threadIdx.x & 31;
    int b  = gw >> 12;          // 16*256 = 4096 wave-blocks per batch
    int mg = (gw >> 8) & 15;    // 64-row M group
    int nt = gw & 255;          // N tile
    int l16    = lane & 15;
    int koff8  = (lane < 16) ? 0 : 8;
    int koff16 = (lane < 16) ? 0 : 16;

    int mbase = mg * 64;        // never straddles the 256/512 boundaries
    const float* W; const float* bias; int mlocal;
    if (mbase < 256)      { W = Wf; bias = bf; mlocal = mbase; }
    else if (mbase < 512) { W = Wg; bias = bg; mlocal = mbase - 256; }
    else                  { W = Wh; bias = bh; mlocal = mbase - 512; }

    const float* Xb = x + (size_t)b * C * N;
    int n = nt * 16 + l16;      // output column (pixel)

    v8f acc[4] = {};
    for (int k0 = 0; k0 < C; k0 += 32) {
        // B frag: X[k0+koff8+i+(i>=8?8:0), n], i = 0..15 (column gather, shared by 4 tiles)
        v16h bb;
        #pragma unroll
        for (int i = 0; i < 16; ++i)
            bb[i] = (_Float16)Xb[(size_t)(k0 + koff8 + i + (i >= 8 ? 8 : 0)) * N + n];

        #pragma unroll
        for (int t = 0; t < 4; ++t) {
            int mrow = mlocal + t * 16 + l16;
            const float* pa = W + (size_t)mrow * C + k0 + koff8;
            v8f f0 = *(const v8f*)pa;
            v8f f1 = *(const v8f*)(pa + 16);
            AFrag a;
            v8h h0, h1;
            #pragma unroll
            for (int i = 0; i < 8; ++i) { h0[i] = (_Float16)f0[i]; h1[i] = (_Float16)f1[i]; }
            a.h[0] = h0; a.h[1] = h1;
            acc[t] = wmma_f16(a.v, bb, acc[t]);
        }
    }

    // C/D layout: VGPR r holds (M = r + koff8, Ncol = l16)
    #pragma unroll
    for (int t = 0; t < 4; ++t) {
        int mt = mlocal + t * 16;
        if (mbase < 512) {
            _Float16* T = ((mbase < 256) ? Ft : Gt) + (size_t)b * N * HC;
            v8h out;
            #pragma unroll
            for (int r = 0; r < 8; ++r)
                out[r] = (_Float16)(acc[t][r] + bias[mt + koff8 + r]);
            *(v8h*)(T + (size_t)n * HC + mt + koff8) = out;   // transposed, contiguous
        } else {
            _Float16* Hp = Hm + (size_t)b * 512 * N;
            #pragma unroll
            for (int r = 0; r < 8; ++r)
                Hp[(size_t)(mt + koff8 + r) * N + n] =
                    (_Float16)(acc[t][r] + bias[mt + koff8 + r]);
        }
    }
}

// ---------------------------------------------------------------------------
// K2: energy + softmax + attention column-mean accumulation, one batch.
// Block = 256 threads (8 waves) owns 16 query rows x 4096 keys.
// A-fragments hoisted into 64 VGPRs, reused across all 256 key tiles.
// LDS: 16*4096 f32 (256 KB < 320 KB WGP).
// ---------------------------------------------------------------------------
__global__ void attn_softmax_kernel(const _Float16* __restrict__ Ft,
                                    const _Float16* __restrict__ Gt,
                                    _Float16* __restrict__ P,
                                    float* __restrict__ att,   // [4096], pre-zeroed
                                    int b)
{
    extern __shared__ float lds[];          // [16][4096]
    const int N = 4096, HC = 256;
    const _Float16* F = Ft + (size_t)b * N * HC;
    const _Float16* G = Gt + (size_t)b * N * HC;

    int tid  = threadIdx.x;
    int wave = tid >> 5, lane = tid & 31;
    int l16    = lane & 15;
    int koff8  = (lane < 16) ? 0 : 8;
    int koff16 = (lane < 16) ? 0 : 16;
    int rowblock = blockIdx.x;              // 0..255
    int nrow = rowblock * 16 + l16;         // query row for A-frag

    const _Float16* Fr = F + (size_t)nrow * HC;
    AFrag a[8];
    #pragma unroll
    for (int kk = 0; kk < 8; ++kk) {
        a[kk].h[0] = *(const v8h*)(Fr + kk * 32 + koff8);
        a[kk].h[1] = *(const v8h*)(Fr + kk * 32 + 16 + koff8);
    }

    // Phase 1: E[16 x 4096] tiles into LDS
    for (int j = 0; j < 32; ++j) {
        int ct   = wave + 8 * j;            // key column tile
        const _Float16* Gr = G + (size_t)(ct * 16 + l16) * HC;
        v8f acc = {};
        #pragma unroll
        for (int kk = 0; kk < 8; ++kk) {
            AFrag bb;
            bb.h[0] = *(const v8h*)(Gr + kk * 32 + koff8);
            bb.h[1] = *(const v8h*)(Gr + kk * 32 + 16 + koff8);
            acc = wmma_f16(a[kk].v, bb.v, acc);
        }
        #pragma unroll
        for (int r = 0; r < 8; ++r)
            lds[(r + koff8) * N + ct * 16 + l16] = acc[r];
    }
    __syncthreads();

    // Phase 2: row softmax (16 threads per row, width-16 shuffle reductions)
    int row = tid >> 4;                     // 0..15
    int c16 = tid & 15;
    float* rowp = lds + row * N;

    float mx = -3.4e38f;
    for (int i = c16; i < N; i += 16) mx = fmaxf(mx, rowp[i]);
    #pragma unroll
    for (int s = 1; s < 16; s <<= 1) mx = fmaxf(mx, __shfl_xor(mx, s, 16));

    float sum = 0.f;
    for (int i = c16; i < N; i += 16) {
        float e = __expf(rowp[i] - mx);
        rowp[i] = e;
        sum += e;
    }
    #pragma unroll
    for (int s = 1; s < 16; s <<= 1) sum += __shfl_xor(sum, s, 16);
    float inv = 1.0f / sum;

    _Float16* Prow = P + (size_t)(rowblock * 16 + row) * N;
    for (int i = c16; i < N; i += 16) {
        float p = rowp[i] * inv;
        rowp[i] = p;
        Prow[i] = (_Float16)p;
    }
    __syncthreads();

    // Phase 3: attention column partial sums (mean over queries)
    const float scale = 1.0f / 4096.0f;
    for (int c = tid; c < N; c += 256) {
        float s = 0.f;
        #pragma unroll
        for (int r = 0; r < 16; ++r) s += lds[r * N + c];
        atomicAdd(att + c, s * scale);
    }
}

// ---------------------------------------------------------------------------
// K3: residual R = Hm(512x4096) * P^T(4096x4096), one batch. Dominant GEMM.
// Block = 4 waves sharing a 4-tile B panel of P staged in LDS via async
// global->LDS loads (double-buffered, ASYNCcnt). Waves: 2 M-tiles x 4 N-tiles
// each -> 8 WMMAs per k-step from LDS-resident B fragments.
// Grid: 256 blocks x 128 threads (bm 0..3 x bn 0..63).
// ---------------------------------------------------------------------------
__global__ void residual_kernel(const _Float16* __restrict__ Hm,
                                const _Float16* __restrict__ P,
                                float* __restrict__ out, int b)
{
    const int N = 4096, C = 512;
    // [buf][64 rows][32 halves] = 4 KB per buffer
    __shared__ __align__(64) _Float16 stage[2][64 * 32];

    int tid  = threadIdx.x;
    int wave = tid >> 5, lane = tid & 31;
    int l16    = lane & 15;
    int koff8  = (lane < 16) ? 0 : 8;
    int koff16 = (lane < 16) ? 0 : 16;
    int bm  = blockIdx.x >> 6;      // 0..3  -> 8 M-tiles per block
    int bn  = blockIdx.x & 63;      // 0..63 -> 4 shared N-tiles per block
    int mts = bm * 8 + wave * 2;    // this wave's first M-tile
    int nts = bn * 4;               // block's first N-tile

    const _Float16* Hp  = Hm + (size_t)b * C * N;
    const _Float16* Ha0 = Hp + (size_t)(mts * 16 + l16) * N;
    const _Float16* Ha1 = Ha0 + (size_t)16 * N;

    // Cooperative staging: 64 P rows x 64 bytes = 256 x 16B chunks, 2 per thread
    // chunk e -> row = e >> 2, seg = e & 3 (seg*16 bytes within the row)
    const _Float16* Pbase = P + (size_t)(nts * 16) * N;
    int e0 = tid;        int s0 = e0 & 3, r0 = e0 >> 2;
    int e1 = tid + 128;  int s1 = e1 & 3, r1 = e1 >> 2;

    auto issue = [&](int k0, int buf) {
#if HAVE_ASYNC_LDS
        async_copy16(Pbase + (size_t)r0 * N + k0 + s0 * 8,
                     &stage[buf][r0 * 32 + s0 * 8]);
        async_copy16(Pbase + (size_t)r1 * N + k0 + s1 * 8,
                     &stage[buf][r1 * 32 + s1 * 8]);
#else
        *(v8h*)&stage[buf][r0 * 32 + s0 * 8] =
            *(const v8h*)(Pbase + (size_t)r0 * N + k0 + s0 * 8);
        *(v8h*)&stage[buf][r1 * 32 + s1 * 8] =
            *(const v8h*)(Pbase + (size_t)r1 * N + k0 + s1 * 8);
#endif
    };

    int buf = 0;
    issue(0, 0);
    wait_async0();
    __syncthreads();

    v8f acc[2][4] = {};
    for (int k0 = 0; k0 < N; k0 += 32) {
        if (k0 + 32 < N) issue(k0 + 32, buf ^ 1);   // prefetch next panel

        AFrag a0, a1;
        a0.h[0] = *(const v8h*)(Ha0 + k0 + koff8);
        a0.h[1] = *(const v8h*)(Ha0 + k0 + 16 + koff8);
        a1.h[0] = *(const v8h*)(Ha1 + k0 + koff8);
        a1.h[1] = *(const v8h*)(Ha1 + k0 + 16 + koff8);
        if (k0 + 512 < N) {
            __builtin_prefetch(Ha0 + k0 + 512, 0, 1);
            __builtin_prefetch(Ha1 + k0 + 512, 0, 1);
        }

        // B fragments from the LDS-resident panel
        AFrag b0, b1, b2, b3;
        b0.h[0] = *(const v8h*)&stage[buf][( 0 + l16) * 32 + koff8];
        b0.h[1] = *(const v8h*)&stage[buf][( 0 + l16) * 32 + 16 + koff8];
        b1.h[0] = *(const v8h*)&stage[buf][(16 + l16) * 32 + koff8];
        b1.h[1] = *(const v8h*)&stage[buf][(16 + l16) * 32 + 16 + koff8];
        b2.h[0] = *(const v8h*)&stage[buf][(32 + l16) * 32 + koff8];
        b2.h[1] = *(const v8h*)&stage[buf][(32 + l16) * 32 + 16 + koff8];
        b3.h[0] = *(const v8h*)&stage[buf][(48 + l16) * 32 + koff8];
        b3.h[1] = *(const v8h*)&stage[buf][(48 + l16) * 32 + 16 + koff8];

        acc[0][0] = wmma_f16(a0.v, b0.v, acc[0][0]);
        acc[1][0] = wmma_f16(a1.v, b0.v, acc[1][0]);
        acc[0][1] = wmma_f16(a0.v, b1.v, acc[0][1]);
        acc[1][1] = wmma_f16(a1.v, b1.v, acc[1][1]);
        acc[0][2] = wmma_f16(a0.v, b2.v, acc[0][2]);
        acc[1][2] = wmma_f16(a1.v, b2.v, acc[1][2]);
        acc[0][3] = wmma_f16(a0.v, b3.v, acc[0][3]);
        acc[1][3] = wmma_f16(a1.v, b3.v, acc[1][3]);

        wait_async0();          // our next-panel writes have landed in LDS
        __syncthreads();        // everyone done reading buf & writing buf^1
        buf ^= 1;
    }

    float* R = out + (size_t)b * C * N;
    #pragma unroll
    for (int t = 0; t < 2; ++t)
        #pragma unroll
        for (int c4 = 0; c4 < 4; ++c4)
            #pragma unroll
            for (int r = 0; r < 8; ++r)
                R[(size_t)((mts + t) * 16 + r + koff8) * N + (nts + c4) * 16 + l16] =
                    acc[t][c4][r];
}

// ---------------------------------------------------------------------------
extern "C" void kernel_launch(void* const* d_in, const int* in_sizes, int n_in,
                              void* d_out, int out_size, void* d_ws, size_t ws_size,
                              hipStream_t stream) {
    (void)in_sizes; (void)n_in; (void)out_size; (void)ws_size;
    const float* x  = (const float*)d_in[0];
    const float* Wf = (const float*)d_in[1];
    const float* bf = (const float*)d_in[2];
    const float* Wg = (const float*)d_in[3];
    const float* bg = (const float*)d_in[4];
    const float* Wh = (const float*)d_in[5];
    const float* bh = (const float*)d_in[6];
    float* outp = (float*)d_out;

    const int B = 4, C = 512, N = 4096, HC = 256;

    // Workspace (f16): Ft[B*N*HC] | Gt[B*N*HC] | Hm[B*C*N] | P[N*N] (reused/batch)
    _Float16* Ft = (_Float16*)d_ws;
    _Float16* Gt = Ft + (size_t)B * N * HC;
    _Float16* Hm = Gt + (size_t)B * N * HC;
    _Float16* P  = Hm + (size_t)B * C * N;

    float* att = outp + (size_t)B * C * N;     // attention after residual
    (void)hipMemsetAsync(att, 0, (size_t)B * N * sizeof(float), stream);

    // K1: all batches; 4 M-tiles per wave -> B*16*256 = 16384 waves
    proj_kernel<<<(B * 16 * 256) / 4, 128, 0, stream>>>(x, Wf, bf, Wg, bg, Wh, bh,
                                                        Ft, Gt, Hm);

    // K2 + K3 per batch (P reused, stream-ordered)
    for (int b = 0; b < B; ++b) {
        attn_softmax_kernel<<<N / 16, 256, 16 * N * sizeof(float), stream>>>(
            Ft, Gt, P, att + (size_t)b * N, b);
        residual_kernel<<<256, 128, 0, stream>>>(Hm, P, outp, b);
    }
}
